// WanSelfAttention_74105365725236
// MI455X (gfx1250) — compile-verified, matrix operand order checked by reading, not run on hardware
//
#include <hip/hip_runtime.h>

typedef __attribute__((ext_vector_type(16))) _Float16 v16h;
typedef __attribute__((ext_vector_type(8)))  _Float16 v8h;
typedef __attribute__((ext_vector_type(8)))  float    v8f;
typedef __attribute__((ext_vector_type(4)))  int      v4i;

#define BB 2
#define SS 2048
#define DD 2048
#define NH 16
#define HD 128
#define MM (BB*SS)      /* 4096 rows */
#define N1 (3*DD)       /* 6144 fused QKV cols */

/* ---- workspace layout (bytes) ---- */
#define OFF_XH   ((size_t)0)
#define OFF_WQKV (OFF_XH   + (size_t)MM*DD*2)     /* x in f16            */
#define OFF_WO   (OFF_WQKV + (size_t)N1*DD*2)     /* [wq;wk;wv] f16      */
#define OFF_Y    (OFF_WO   + (size_t)DD*DD*2)     /* wo f16              */
#define OFF_QH   (OFF_Y    + (size_t)MM*N1*4)     /* fused QKV f32       */
#define OFF_KH   (OFF_QH   + (size_t)MM*DD*2)     /* q f16 [B,N,S,HD]    */
#define OFF_VH   (OFF_KH   + (size_t)MM*DD*2)
#define OFF_OH   (OFF_VH   + (size_t)MM*DD*2)     /* attn out f16 [B,S,D]*/

/* ---- CDNA5 async global->LDS copy (ASYNCcnt-tracked, bypasses VGPRs) ---- */
#if __has_builtin(__builtin_amdgcn_global_load_async_to_lds_b128) && \
    __has_builtin(__builtin_amdgcn_s_wait_asynccnt)
#define USE_ASYNC 1
#else
#define USE_ASYNC 0
#endif

/* b128 payload type: <4 x i32>; clang prints AS(1) as `__device__` in HIP */
typedef __attribute__((address_space(1))) v4i gv4i;
typedef __attribute__((address_space(3))) v4i lv4i;

__device__ __forceinline__ void async_cp16(const void* g, void* l) {
#if USE_ASYNC
  /* generic LDS pointers carry the LDS byte offset in the low 32 bits
     (flat->LDS aperture mapping) */
  __builtin_amdgcn_global_load_async_to_lds_b128(
      (gv4i*)(unsigned long long)(size_t)g,
      (lv4i*)(unsigned int)(size_t)l, 0, 0);
#else
  (void)g; (void)l;
#endif
}

/* ---------------- f32 -> f16 convert ---------------- */
__global__ __launch_bounds__(256) void cvt_f16(const float* __restrict__ src,
                                               _Float16* __restrict__ dst, int n) {
  for (int i = blockIdx.x * 256 + threadIdx.x; i < n; i += gridDim.x * 256)
    dst[i] = (_Float16)src[i];
}

/* Fragment loader per CDNA5 ISA 7.12.2 (16-bit A 16x32 / B 32x16 layouts).
   tile: pointer to the 16-row tile origin; row dim stride = `stride` halves;
   contiguous (matmul-K) dim carries the 8+8 half split at +0 / +16. */
__device__ __forceinline__ v16h ld_frag(const _Float16* tile, int stride) {
  const int lane = threadIdx.x & 31;
  const _Float16* p = tile + (size_t)(lane & 15) * stride + ((lane >> 4) << 3);
  v8h lo = *(const v8h*)(p);
  v8h hi = *(const v8h*)(p + 16);
  v16h r;
#pragma unroll
  for (int i = 0; i < 8; ++i) { r[i] = lo[i]; r[8 + i] = hi[i]; }
  return r;
}

/* ---------------- WMMA GEMM: C[M,N] = A[M,K] * Bw[N,K]^T + bias ----------------
   128x128 block tile, 8 waves, wave tile 32x64 (2x4 of 16x16), K-step 32.
   Global->LDS staging via async copy engine, LDS double buffered:
   4 async b128 ops per wave per stage; s_wait_asynccnt 4 retires the previous
   stage (per-wave in-order completion) while the next stage streams. */
__global__ __launch_bounds__(256) void gemm_wmma(
    const _Float16* __restrict__ A, const _Float16* __restrict__ Bw,
    float* __restrict__ C, int Mdim, int Ndim, int Kdim,
    const float* __restrict__ b0, const float* __restrict__ b1,
    const float* __restrict__ b2, int s1, int s2)
{
  __shared__ __align__(16) _Float16 ldsA[2][128 * 32];
  __shared__ __align__(16) _Float16 ldsB[2][128 * 32];
  const int tid  = threadIdx.x;
  const int lane = tid & 31, wave = tid >> 5;
  const int l15  = lane & 15, hi8 = (lane >> 4) << 3;
  const int wm = wave >> 1, wn = wave & 1;
  const int mBase = blockIdx.y * 128;
  const int nBase = blockIdx.x * 128;

  const v8f zf = {};
  v8f acc[2][4];
#pragma unroll
  for (int i = 0; i < 2; i++)
#pragma unroll
    for (int j = 0; j < 4; j++) acc[i][j] = zf;

  int rowT[2], segT[2];
#pragma unroll
  for (int t = 0; t < 2; t++) {
    int c = tid + t * 256;            /* 512 16B chunks per 128x32 tile */
    rowT[t] = c >> 2;
    segT[t] = (c & 3) << 3;
  }

  const int nk = Kdim >> 5;

#if USE_ASYNC
  /* ---- async pipeline ---- */
#pragma unroll
  for (int t = 0; t < 2; t++) {       /* stage 0 */
    async_cp16(A  + (size_t)(mBase + rowT[t]) * Kdim + segT[t],
               &ldsA[0][rowT[t] * 32 + segT[t]]);
    async_cp16(Bw + (size_t)(nBase + rowT[t]) * Kdim + segT[t],
               &ldsB[0][rowT[t] * 32 + segT[t]]);
  }
  for (int kb = 0; kb < nk; ++kb) {
    const int st = kb & 1;
    if (kb + 1 < nk) {
      const int k0 = (kb + 1) << 5;
      __builtin_prefetch(A + (size_t)(mBase + rowT[0]) * Kdim + k0 + 64, 0, 0);
#pragma unroll
      for (int t = 0; t < 2; t++) {
        async_cp16(A  + (size_t)(mBase + rowT[t]) * Kdim + k0 + segT[t],
                   &ldsA[st ^ 1][rowT[t] * 32 + segT[t]]);
        async_cp16(Bw + (size_t)(nBase + rowT[t]) * Kdim + k0 + segT[t],
                   &ldsB[st ^ 1][rowT[t] * 32 + segT[t]]);
      }
      __builtin_amdgcn_s_wait_asynccnt(4);   /* previous stage landed */
    } else {
      __builtin_amdgcn_s_wait_asynccnt(0);
    }
    __syncthreads();
    v16h af[2], bf[4];
#pragma unroll
    for (int i = 0; i < 2; i++) af[i] = ld_frag(&ldsA[st][(wm * 32 + i * 16) * 32], 32);
#pragma unroll
    for (int j = 0; j < 4; j++) bf[j] = ld_frag(&ldsB[st][(wn * 64 + j * 16) * 32], 32);
#pragma unroll
    for (int i = 0; i < 2; i++)
#pragma unroll
      for (int j = 0; j < 4; j++)
        acc[i][j] = __builtin_amdgcn_wmma_f32_16x16x32_f16(
            false, af[i], false, bf[j], (short)0, acc[i][j], false, false);
    __syncthreads();
  }
#else
  /* ---- fallback: register-staged pipeline ---- */
  v8h ra[2], rb[2];
#pragma unroll
  for (int t = 0; t < 2; t++) {
    ra[t] = *(const v8h*)(A  + (size_t)(mBase + rowT[t]) * Kdim + segT[t]);
    rb[t] = *(const v8h*)(Bw + (size_t)(nBase + rowT[t]) * Kdim + segT[t]);
  }
  for (int kb = 0; kb < nk; ++kb) {
#pragma unroll
    for (int t = 0; t < 2; t++) {
      *(v8h*)(&ldsA[0][rowT[t] * 32 + segT[t]]) = ra[t];
      *(v8h*)(&ldsB[0][rowT[t] * 32 + segT[t]]) = rb[t];
    }
    __syncthreads();
    if (kb + 1 < nk) {
      const int k0 = (kb + 1) << 5;
      __builtin_prefetch(A + (size_t)(mBase + rowT[0]) * Kdim + k0 + 32, 0, 0);
#pragma unroll
      for (int t = 0; t < 2; t++) {
        ra[t] = *(const v8h*)(A  + (size_t)(mBase + rowT[t]) * Kdim + k0 + segT[t]);
        rb[t] = *(const v8h*)(Bw + (size_t)(nBase + rowT[t]) * Kdim + k0 + segT[t]);
      }
    }
    v16h af[2], bf[4];
#pragma unroll
    for (int i = 0; i < 2; i++) af[i] = ld_frag(&ldsA[0][(wm * 32 + i * 16) * 32], 32);
#pragma unroll
    for (int j = 0; j < 4; j++) bf[j] = ld_frag(&ldsB[0][(wn * 64 + j * 16) * 32], 32);
#pragma unroll
    for (int i = 0; i < 2; i++)
#pragma unroll
      for (int j = 0; j < 4; j++)
        acc[i][j] = __builtin_amdgcn_wmma_f32_16x16x32_f16(
            false, af[i], false, bf[j], (short)0, acc[i][j], false, false);
    __syncthreads();
  }
#endif

  /* epilogue: C/D layout — lane holds col, VGPR r holds row r + 8*(lane>=16) */
#pragma unroll
  for (int i = 0; i < 2; i++)
#pragma unroll
    for (int j = 0; j < 4; j++) {
      const int col = nBase + wn * 64 + j * 16 + l15;
      const float bias = (col < s1) ? b0[col] : (col < s2) ? b1[col - s1] : b2[col - s2];
#pragma unroll
      for (int r = 0; r < 8; r++) {
        const int row = mBase + wm * 32 + i * 16 + r + hi8;
        C[(size_t)row * Ndim + col] = acc[i][j][r] + bias;
      }
    }
}

/* ---------------- RMSNorm + RoPE + repack to [B,N,S,HD] f16 ---------------- */
__global__ __launch_bounds__(256) void norm_rope(
    const float* __restrict__ Y, const float* __restrict__ freqs,
    const float* __restrict__ gq, const float* __restrict__ gk,
    _Float16* __restrict__ qh, _Float16* __restrict__ kh, _Float16* __restrict__ vh)
{
  const int m = blockIdx.x;
  const int b = m / SS, s = m % SS;
  const int tid = threadIdx.x;
  const float* Yr = Y + (size_t)m * N1;
  const int c0 = tid * 8;

  float qv[8], kv[8], vv[8];
  float sq = 0.f, sk = 0.f;
#pragma unroll
  for (int j = 0; j < 8; j++) {
    qv[j] = Yr[c0 + j]; kv[j] = Yr[DD + c0 + j]; vv[j] = Yr[2 * DD + c0 + j];
    sq += qv[j] * qv[j]; sk += kv[j] * kv[j];
  }
  for (int msk = 16; msk; msk >>= 1) {
    sq += __shfl_xor(sq, msk, 32);
    sk += __shfl_xor(sk, msk, 32);
  }
  __shared__ float red[16];
  const int wid = tid >> 5;
  if ((tid & 31) == 0) { red[wid] = sq; red[8 + wid] = sk; }
  __syncthreads();
  if (tid == 0) {
    float a = 0.f, c = 0.f;
    for (int i = 0; i < 8; i++) { a += red[i]; c += red[8 + i]; }
    red[0] = a; red[8] = c;
  }
  __syncthreads();
  const float rq = rsqrtf(red[0] * (1.0f / DD) + 1e-6f);
  const float rk = rsqrtf(red[8] * (1.0f / DD) + 1e-6f);

  const int n  = c0 / HD;
  const int d0 = c0 % HD;
  const size_t ob = ((size_t)(b * NH + n) * SS + s) * HD + d0;
#pragma unroll
  for (int j = 0; j < 8; j += 2) {
    const int p = (d0 + j) >> 1;
    const float f  = freqs[(size_t)s * (HD / 2) + p];
    const float cf = __cosf(f), sf = __sinf(f);
    const float u0 = qv[j]     * rq * gq[c0 + j];
    const float u1 = qv[j + 1] * rq * gq[c0 + j + 1];
    qh[ob + j]     = (_Float16)(u0 * cf - u1 * sf);
    qh[ob + j + 1] = (_Float16)(u0 * sf + u1 * cf);
    const float w0 = kv[j]     * rk * gk[c0 + j];
    const float w1 = kv[j + 1] * rk * gk[c0 + j + 1];
    kh[ob + j]     = (_Float16)(w0 * cf - w1 * sf);
    kh[ob + j + 1] = (_Float16)(w0 * sf + w1 * cf);
    vh[ob + j]     = (_Float16)vv[j];
    vh[ob + j + 1] = (_Float16)vv[j + 1];
  }
}

/* ---------------- Flash attention (WMMA QK^T and PV) ----------------
   128 threads = 4 waves; each wave owns 32 query rows; Q kept in registers;
   key tiles of 32 streamed through LDS: K via async copy engine (ASYNCcnt),
   V transposed through VGPRs (layout change). */
__global__ __launch_bounds__(128) void attn_wmma(
    const _Float16* __restrict__ qh, const _Float16* __restrict__ kh,
    const _Float16* __restrict__ vh, const int* __restrict__ seq_lens,
    _Float16* __restrict__ oh)
{
  __shared__ __align__(16) _Float16 ldsK[32 * 128];
  __shared__ __align__(16) _Float16 ldsVt[128 * 32];
  __shared__ __align__(16) _Float16 ldsP[4 * 32 * 32];

  const int tid = threadIdx.x;
  const int lane = tid & 31, w = tid >> 5;
  const int l15 = lane & 15, hi8 = (lane >> 4) << 3;
  const int nqt = SS / 128;
  const int qt = blockIdx.x % nqt;
  const int h  = (blockIdx.x / nqt) % NH;
  const int b  = blockIdx.x / (nqt * NH);
  const int slen = seq_lens[b];
  const float SCALE = 0.08838834764831845f; /* 1/sqrt(128) */

  const size_t headO = (size_t)(b * NH + h) * SS * HD;
  const _Float16* qbase = qh + headO + (size_t)(qt * 128 + w * 32) * HD;
  const _Float16* kbase = kh + headO;
  const _Float16* vbase = vh + headO;

  v16h qf[2][4];
#pragma unroll
  for (int i = 0; i < 2; i++)
#pragma unroll
    for (int dc = 0; dc < 4; dc++)
      qf[i][dc] = ld_frag(qbase + (size_t)(i * 16) * HD + dc * 32, HD);

  const v8f zf = {};
  v8f o[2][8];
#pragma unroll
  for (int i = 0; i < 2; i++)
#pragma unroll
    for (int dj = 0; dj < 8; dj++) o[i][dj] = zf;
  float mrow[2][8], lrow[2][8];
#pragma unroll
  for (int i = 0; i < 2; i++)
#pragma unroll
    for (int r = 0; r < 8; r++) { mrow[i][r] = -3.0e38f; lrow[i][r] = 0.f; }

  const int nkt = (slen + 31) >> 5;
  for (int kt = 0; kt < nkt; ++kt) {
    /* stage K tile [32][128] via async engine; V transposed to [128][32] */
#pragma unroll
    for (int t = 0; t < 4; t++) {
      const int c = tid + t * 128;            /* 512 chunks of 8 halves */
      const int kk = c >> 4, seg = (c & 15) << 3;
#if USE_ASYNC
      async_cp16(kbase + (size_t)(kt * 32 + kk) * HD + seg, ldsK + kk * HD + seg);
#else
      v8h kv = *(const v8h*)(kbase + (size_t)(kt * 32 + kk) * HD + seg);
      *(v8h*)(ldsK + kk * HD + seg) = kv;
#endif
      v8h vv = *(const v8h*)(vbase + (size_t)(kt * 32 + kk) * HD + seg);
#pragma unroll
      for (int e = 0; e < 8; e++) ldsVt[(seg + e) * 32 + kk] = vv[e];
    }
#if USE_ASYNC
    __builtin_amdgcn_s_wait_asynccnt(0);
#endif
    __syncthreads();

    /* scores = Q * K^T : A=Q rows, B=K rows (both K-contiguous over d) */
    v8f sc[2][2];
#pragma unroll
    for (int i = 0; i < 2; i++)
#pragma unroll
      for (int j = 0; j < 2; j++) sc[i][j] = zf;
#pragma unroll
    for (int dc = 0; dc < 4; dc++) {
      v16h bk0 = ld_frag(ldsK + 0 * 16 * HD + dc * 32, HD);
      v16h bk1 = ld_frag(ldsK + 1 * 16 * HD + dc * 32, HD);
#pragma unroll
      for (int i = 0; i < 2; i++) {
        sc[i][0] = __builtin_amdgcn_wmma_f32_16x16x32_f16(
            false, qf[i][dc], false, bk0, (short)0, sc[i][0], false, false);
        sc[i][1] = __builtin_amdgcn_wmma_f32_16x16x32_f16(
            false, qf[i][dc], false, bk1, (short)0, sc[i][1], false, false);
      }
    }

    /* scale + key mask (lane owns one key column) */
#pragma unroll
    for (int j = 0; j < 2; j++) {
      const int kg = kt * 32 + j * 16 + l15;
      const bool valid = kg < slen;
#pragma unroll
      for (int i = 0; i < 2; i++)
#pragma unroll
        for (int r = 0; r < 8; r++)
          sc[i][j][r] = valid ? sc[i][j][r] * SCALE : -3.0e38f;
    }

    /* online softmax: row stats live across the 16 lanes of a half-wave */
    float alr[2][8];
#pragma unroll
    for (int i = 0; i < 2; i++)
#pragma unroll
      for (int r = 0; r < 8; r++) {
        float t = fmaxf(sc[i][0][r], sc[i][1][r]);
        for (int msk = 8; msk; msk >>= 1) t = fmaxf(t, __shfl_xor(t, msk, 32));
        const float mn = fmaxf(mrow[i][r], t);
        const float al = __expf(mrow[i][r] - mn);
        const float p0 = __expf(sc[i][0][r] - mn);
        const float p1 = __expf(sc[i][1][r] - mn);
        sc[i][0][r] = p0; sc[i][1][r] = p1;
        float rs = p0 + p1;
        for (int msk = 8; msk; msk >>= 1) rs += __shfl_xor(rs, msk, 32);
        lrow[i][r] = lrow[i][r] * al + rs;
        mrow[i][r] = mn;
        alr[i][r]  = al;
      }
#pragma unroll
    for (int i = 0; i < 2; i++)
#pragma unroll
      for (int dj = 0; dj < 8; dj++)
#pragma unroll
        for (int r = 0; r < 8; r++) o[i][dj][r] *= alr[i][r];

    /* P: C-layout -> A-layout via per-wave LDS patch */
    _Float16* Pw = ldsP + w * 32 * 32;
#pragma unroll
    for (int i = 0; i < 2; i++)
#pragma unroll
      for (int j = 0; j < 2; j++)
#pragma unroll
        for (int r = 0; r < 8; r++)
          Pw[(i * 16 + r + hi8) * 32 + j * 16 + l15] = (_Float16)sc[i][j][r];
    __builtin_amdgcn_wave_barrier();

    /* O += P * V  (B frags from transposed V: N=d, matmul-K=key) */
    v16h pf0 = ld_frag(Pw, 32);
    v16h pf1 = ld_frag(Pw + 16 * 32, 32);
#pragma unroll
    for (int dj = 0; dj < 8; dj++) {
      v16h vf = ld_frag(ldsVt + dj * 16 * 32, 32);
      o[0][dj] = __builtin_amdgcn_wmma_f32_16x16x32_f16(
          false, pf0, false, vf, (short)0, o[0][dj], false, false);
      o[1][dj] = __builtin_amdgcn_wmma_f32_16x16x32_f16(
          false, pf1, false, vf, (short)0, o[1][dj], false, false);
    }
    __syncthreads();
  }

  /* normalize and write [B,S,D] f16 for the output projection */
#pragma unroll
  for (int i = 0; i < 2; i++)
#pragma unroll
    for (int r = 0; r < 8; r++) {
      const int srow = qt * 128 + w * 32 + i * 16 + r + hi8;
      const float inv = 1.0f / lrow[i][r];
#pragma unroll
      for (int dj = 0; dj < 8; dj++) {
        const int col = h * HD + dj * 16 + l15;
        oh[((size_t)(b * SS + srow)) * DD + col] = (_Float16)(o[i][dj][r] * inv);
      }
    }
}

/* ---------------- host-side orchestration ---------------- */
extern "C" void kernel_launch(void* const* d_in, const int* in_sizes, int n_in,
                              void* d_out, int out_size, void* d_ws, size_t ws_size,
                              hipStream_t stream) {
  (void)in_sizes; (void)n_in; (void)out_size; (void)ws_size;
  const float* x        = (const float*)d_in[0];
  const int*   seq_lens = (const int*)d_in[1];
  const float* freqs    = (const float*)d_in[2];
  const float* wq = (const float*)d_in[3];
  const float* bq = (const float*)d_in[4];
  const float* wk = (const float*)d_in[5];
  const float* bk = (const float*)d_in[6];
  const float* wv = (const float*)d_in[7];
  const float* bv = (const float*)d_in[8];
  const float* wo = (const float*)d_in[9];
  const float* bo = (const float*)d_in[10];
  const float* gq = (const float*)d_in[11];
  const float* gk = (const float*)d_in[12];

  char* ws = (char*)d_ws;
  _Float16* xh    = (_Float16*)(ws + OFF_XH);
  _Float16* wqkvh = (_Float16*)(ws + OFF_WQKV);
  _Float16* woh   = (_Float16*)(ws + OFF_WO);
  float*    Y     = (float*)(ws + OFF_Y);
  _Float16* qhp   = (_Float16*)(ws + OFF_QH);
  _Float16* khp   = (_Float16*)(ws + OFF_KH);
  _Float16* vhp   = (_Float16*)(ws + OFF_VH);
  _Float16* ohp   = (_Float16*)(ws + OFF_OH);

  /* 1) down-convert activations + weights to f16 */
  cvt_f16<<<1024, 256, 0, stream>>>(x,  xh,               MM * DD);
  cvt_f16<<<1024, 256, 0, stream>>>(wq, wqkvh,            DD * DD);
  cvt_f16<<<1024, 256, 0, stream>>>(wk, wqkvh + (size_t)DD * DD,     DD * DD);
  cvt_f16<<<1024, 256, 0, stream>>>(wv, wqkvh + (size_t)2 * DD * DD, DD * DD);
  cvt_f16<<<1024, 256, 0, stream>>>(wo, woh,              DD * DD);

  /* 2) fused QKV projection: Y[4096,6144] = xh * [wq;wk;wv]^T + [bq|bk|bv] */
  gemm_wmma<<<dim3(N1 / 128, MM / 128), 256, 0, stream>>>(
      xh, wqkvh, Y, MM, N1, DD, bq, bk, bv, DD, 2 * DD);

  /* 3) RMSNorm + RoPE + repack */
  norm_rope<<<MM, 256, 0, stream>>>(Y, freqs, gq, gk, qhp, khp, vhp);

  /* 4) flash attention */
  attn_wmma<<<BB * NH * (SS / 128), 128, 0, stream>>>(qhp, khp, vhp, seq_lens, ohp);

  /* 5) output projection into d_out (f32) */
  gemm_wmma<<<dim3(DD / 128, MM / 128), 256, 0, stream>>>(
      ohp, woh, (float*)d_out, MM, DD, DD, bo, bo, bo, DD, 2 * DD);
}
